// PerformerAttention_90194313216409
// MI455X (gfx1250) — matrix operand order repaired and checked by reading, not compiled, not run on hardware
//
#include <hip/hip_runtime.h>

// ---------------------------------------------------------------------------
// Performer attention on gfx1250 (MI455X): bf16 WMMA + double-buffered TDM.
// ---------------------------------------------------------------------------

#if __has_include(<hip/amd_detail/amd_gfx1250_TDM.h>)
#define TDM_6ARG 1
#endif

typedef __bf16 bf16;
typedef __attribute__((ext_vector_type(16))) __bf16 v16bf;
typedef __attribute__((ext_vector_type(8)))  __bf16 v8bf;
typedef __attribute__((ext_vector_type(4)))  __bf16 v4bf;
typedef __attribute__((ext_vector_type(2)))  __bf16 v2bf;
typedef __attribute__((ext_vector_type(8)))  float  v8f;
typedef __attribute__((ext_vector_type(4)))  unsigned v4u;
typedef __attribute__((ext_vector_type(8)))  int      v8i;
typedef __attribute__((ext_vector_type(4)))  int      v4i;

__device__ __forceinline__ bf16 f32_to_bf16(float f) {
  unsigned u = __builtin_bit_cast(unsigned, f);
  unsigned r = u + 0x7FFFu + ((u >> 16) & 1u);  // round-nearest-even
  unsigned short h = (unsigned short)(r >> 16);
  return __builtin_bit_cast(bf16, h);
}
__device__ __forceinline__ float bf16_to_f32(bf16 b) {
  unsigned u = ((unsigned)__builtin_bit_cast(unsigned short, b)) << 16;
  return __builtin_bit_cast(float, u);
}
__device__ __forceinline__ unsigned pack2(bf16 lo, bf16 hi) {
  return (unsigned)__builtin_bit_cast(unsigned short, lo) |
         ((unsigned)__builtin_bit_cast(unsigned short, hi) << 16);
}
__device__ __forceinline__ unsigned lds_off_of(const void* p) {
  return (unsigned)(unsigned long long)p;  // LDS aperture: addr[31:0] = offset
}

__device__ __forceinline__ v8f wmma_bf16(v16bf a, v16bf b, v8f c) {
  return __builtin_amdgcn_wmma_f32_16x16x32_bf16(
      false, a, false, b, (short)0, c, false, false);
}

// A fragment (16x32 bf16) from LDS [row][k], row stride `stride`.
__device__ __forceinline__ v16bf frag_a(const bf16* lds, int stride,
                                        int rowbase, int k0, int lane) {
  int row = rowbase + (lane & 15);
  int g = (lane >> 4) & 1;
  const bf16* p = lds + row * stride + k0;
  v16bf a;
#pragma unroll
  for (int j = 0; j < 4; ++j) {
    v2bf lo = *(const v2bf*)(p + g * 8 + 2 * j);
    v2bf hi = *(const v2bf*)(p + 16 + g * 8 + 2 * j);
    a[2 * j]     = lo[0]; a[2 * j + 1]     = lo[1];
    a[8 + 2 * j] = hi[0]; a[8 + 2 * j + 1] = hi[1];
  }
  return a;
}

// B fragment (32x16 bf16) from LDS staged K-major: ldsT[col][k].
__device__ __forceinline__ v16bf frag_b(const bf16* ldsT, int stride,
                                        int colbase, int k0, int lane) {
  int col = colbase + (lane & 15);
  int g = (lane >> 4) & 1;
  const bf16* p = ldsT + col * stride + k0;
  v16bf b;
#pragma unroll
  for (int j = 0; j < 4; ++j) {
    v2bf lo = *(const v2bf*)(p + g * 8 + 2 * j);
    v2bf hi = *(const v2bf*)(p + 16 + g * 8 + 2 * j);
    b[2 * j]     = lo[0]; b[2 * j + 1]     = lo[1];
    b[8 + 2 * j] = hi[0]; b[8 + 2 * j + 1] = hi[1];
  }
  return b;
}

// TDM: DMA a 2D bf16 tile [tile_rows x tile_cols] (row stride stride_elems)
// from global into LDS (rows packed back-to-back, tile_cols per row).
// D# layout per CDNA5 ISA.
__device__ __forceinline__ void tdm_load_tile_2d(
    unsigned lds_off, const bf16* gptr, int tile_cols, int tile_rows,
    int tensor_cols, int tensor_rows, int stride_elems) {
  unsigned long long ga = (unsigned long long)gptr;
  v4u g0;
  g0[0] = 1u;                                   // count=1 valid descriptor
  g0[1] = lds_off;                              // LDS byte address
  g0[2] = (unsigned)ga;                         // global_addr[31:0]
  g0[3] = ((unsigned)(ga >> 32) & 0x01FFFFFFu)  // global_addr[56:32]
          | (2u << 30);                         // type=2 ("image")
  v8i g1;
  g1[0] = (int)(1u << 16);                      // wg_mask=0, data_size=2B
  g1[1] = (int)(((unsigned)tensor_cols & 0xFFFFu) << 16);
  g1[2] = (int)((((unsigned)tensor_cols >> 16) & 0xFFFFu) |
                (((unsigned)tensor_rows & 0xFFFFu) << 16));
  g1[3] = (int)((((unsigned)tensor_rows >> 16) & 0xFFFFu) |
                ((unsigned)tile_cols << 16));
  g1[4] = tile_rows;                            // tile_dim1 (tile_dim2=0)
  g1[5] = stride_elems;                         // tensor_dim0_stride[31:0]
  g1[6] = 0;
  g1[7] = 0;
  v4i gz = {};
#if defined(TDM_6ARG)
  v8i gz8 = {};
  __builtin_amdgcn_tensor_load_to_lds(g0, g1, gz, gz, gz8, 0);
#else
  __builtin_amdgcn_tensor_load_to_lds(g0, g1, gz, gz, 0);
#endif
}

// ---------------------------------------------------------------------------
// f32 -> bf16 cast (plain copy)
// ---------------------------------------------------------------------------
__global__ void cast_f32_bf16_kernel(const float* __restrict__ in,
                                     bf16* __restrict__ out, int n) {
  int i = (blockIdx.x * 256 + threadIdx.x) * 4;
  if (i + 3 < n) {
    float4 v = *(const float4*)(in + i);
    out[i]     = f32_to_bf16(v.x);
    out[i + 1] = f32_to_bf16(v.y);
    out[i + 2] = f32_to_bf16(v.z);
    out[i + 3] = f32_to_bf16(v.w);
  } else {
    for (; i < n; ++i) out[i] = f32_to_bf16(in[i]);
  }
}

// ---------------------------------------------------------------------------
// f32 [K][N] -> bf16 transposed [N][K] (weights become TDM-able B^T tiles)
// ---------------------------------------------------------------------------
__global__ __launch_bounds__(256) void transpose_cast_kernel(
    const float* __restrict__ in, bf16* __restrict__ out, int Kd, int Nd) {
  __shared__ float t[32][33];
  const int tx = threadIdx.x & 31, ty = threadIdx.x >> 5;  // 32 x 8
  const int kb = blockIdx.y * 32, nb = blockIdx.x * 32;
#pragma unroll
  for (int j = 0; j < 4; ++j)
    t[ty + j * 8][tx] = in[(size_t)(kb + ty + j * 8) * Nd + nb + tx];
  __syncthreads();
#pragma unroll
  for (int j = 0; j < 4; ++j)
    out[(size_t)(nb + ty + j * 8) * Kd + kb + tx] =
        f32_to_bf16(t[tx][ty + j * 8]);
}

// ---------------------------------------------------------------------------
// bf16 GEMM: C[MxN] = A[MxK]*Bt[NxK]^T (+bias). Tile 128x64, 8 waves; wave w
// owns 16-row stripe w, four 16-col subtiles (4 WMMA / slab, shared A frag).
// BOTH operands streamed by the Tensor Data Mover into double-buffered LDS:
// slab s+1 DMA overlaps slab s WMMAs.
// ---------------------------------------------------------------------------
__global__ __launch_bounds__(256) void gemm_bf16_kernel(
    const bf16* __restrict__ A, const bf16* __restrict__ Bt,
    int M, int N, int K,
    float* Cf, bf16* Cb, const float* __restrict__ bias) {
  __shared__ bf16 ldsA[2][128 * 32];  // [m][k]
  __shared__ bf16 ldsB[2][64 * 32];   // [n][k]
  const int tid = threadIdx.x, lane = tid & 31, wave = tid >> 5;
  const int m0 = blockIdx.y * 128, n0 = blockIdx.x * 64;
  const int mb = wave * 16;
  v8f acc[4] = {{}, {}, {}, {}};
  const int nsteps = K / 32;
  if (wave == 0) {  // preload slab 0
    tdm_load_tile_2d(lds_off_of(&ldsA[0][0]), A + (size_t)m0 * K,
                     32, 128, K, M, K);
    tdm_load_tile_2d(lds_off_of(&ldsB[0][0]), Bt + (size_t)n0 * K,
                     32, 64, K, N, K);
  }
  for (int s = 0; s < nsteps; ++s) {
    const int cur = s & 1;
    if (wave == 0) __builtin_amdgcn_s_wait_tensorcnt(0);
    __syncthreads();  // slab s visible; prior reads of buffer cur^1 done
    if (wave == 0 && s + 1 < nsteps) {  // DMA slab s+1 overlaps compute
      int k1 = (s + 1) * 32;
      tdm_load_tile_2d(lds_off_of(&ldsA[cur ^ 1][0]),
                       A + (size_t)m0 * K + k1, 32, 128, K, M, K);
      tdm_load_tile_2d(lds_off_of(&ldsB[cur ^ 1][0]),
                       Bt + (size_t)n0 * K + k1, 32, 64, K, N, K);
    }
    const bf16* La = &ldsA[cur][0];
    const bf16* Lb = &ldsB[cur][0];
    v16bf a = frag_a(La, 32, mb, 0, lane);
#pragma unroll
    for (int j = 0; j < 4; ++j) {
      v16bf bf = frag_b(Lb, 32, j * 16, 0, lane);
      acc[j] = wmma_bf16(a, bf, acc[j]);
    }
    __syncthreads();
  }
  const int rowb = m0 + mb + ((lane >> 4) ? 8 : 0);
#pragma unroll
  for (int j = 0; j < 4; ++j) {
    int col = n0 + j * 16 + (lane & 15);
#pragma unroll
    for (int r = 0; r < 8; ++r) {
      float c = acc[j][r];
      if (bias) c += bias[col];
      if (Cf) Cf[(size_t)(rowb + r) * N + col] = c;
      if (Cb) Cb[(size_t)(rowb + r) * N + col] = f32_to_bf16(c);
    }
  }
}

// ---------------------------------------------------------------------------
// phi kernel: proj = Qtile[64x64] * omega^T[64x256] (WMMA),
// phi = exp(proj - 0.5||q||^2)/sqrt(M) -> bf16 [BH*N, 256].
// Q tile and omega both TDM-loaded (omega row-major == B^T staging).
// ---------------------------------------------------------------------------
__global__ __launch_bounds__(256) void phi_kernel(
    const bf16* __restrict__ Qb,   // [B*N, 1024]
    const bf16* __restrict__ Om,   // [256, 64]
    bf16* __restrict__ Phi) {      // [B*H*N, 256]
  __shared__ bf16 ldsQ[64 * 64];
  __shared__ bf16 ldsO[256 * 64];
  __shared__ float ldsNorm[64];
  const int tid = threadIdx.x, lane = tid & 31, wave = tid >> 5;
  const int bid = blockIdx.x;
  const int nt = bid & 63, h = (bid >> 6) & 15, b = bid >> 10;
  const int n0 = nt * 64;
  if (wave == 0) {
    tdm_load_tile_2d(lds_off_of(&ldsQ[0]),
                     Qb + (size_t)(b * 4096 + n0) * 1024 + h * 64,
                     64, 64, 1024, 8192, 1024);
    tdm_load_tile_2d(lds_off_of(&ldsO[0]), Om, 16384, 1, 16384, 1, 16384);
    __builtin_amdgcn_s_wait_tensorcnt(0);
  }
  __syncthreads();
  if (tid < 64) {
    float s = 0.f;
    for (int d = 0; d < 64; ++d) {
      float q = bf16_to_f32(ldsQ[tid * 64 + d]);
      s += q * q;
    }
    ldsNorm[tid] = 0.5f * s;
  }
  __syncthreads();
  const int mb = (wave & 3) * 16;
  const int row = mb + ((lane >> 4) ? 8 : 0);
  const size_t outbase = ((size_t)(b * 16 + h) * 4096 + n0) * 256;
  v16bf a0 = frag_a(ldsQ, 64, mb, 0, lane);
  v16bf a1 = frag_a(ldsQ, 64, mb, 32, lane);
#pragma unroll
  for (int j = 0; j < 8; ++j) {
    int nb = ((wave >> 2) * 8 + j) * 16;
    v16bf b0 = frag_b(ldsO, 64, nb, 0, lane);
    v16bf b1 = frag_b(ldsO, 64, nb, 32, lane);
    v8f acc = {};
    acc = wmma_bf16(a0, b0, acc);
    acc = wmma_bf16(a1, b1, acc);
    int col = nb + (lane & 15);
#pragma unroll
    for (int r = 0; r < 8; ++r) {
      float p = __expf(acc[r] - ldsNorm[row + r]) * 0.0625f;
      Phi[outbase + (size_t)(row + r) * 256 + col] = f32_to_bf16(p);
    }
  }
}

// ---------------------------------------------------------------------------
// kv kernel: per (b,h,mtile=64): kv[64x64] = sum_n phi_k[n,m]^T v[n,d],
// k_sum[m] = sum_n phi_k[n,m]. Needs a transpose -> thread staging
// (pair-packed b32 LDS stores) + global_prefetch_b8 of next slab.
// ---------------------------------------------------------------------------
__global__ __launch_bounds__(256) void kv_kernel(
    const bf16* __restrict__ PhiK,  // [BH*N, 256]
    const bf16* __restrict__ Vb,    // [B*N, 1024]
    float* __restrict__ KV,         // [BH, 256, 64]
    float* __restrict__ Ksum) {     // [BH, 256]
  __shared__ bf16 ldsA[64 * 32];  // phi^T: [m][n]
  __shared__ bf16 ldsB[64 * 32];  // v^T:   [d][n]
  const int tid = threadIdx.x, lane = tid & 31, wave = tid >> 5;
  const int bid = blockIdx.x;
  const int mt = bid & 3, bh = bid >> 2;
  const int h = bh & 15, b = bh >> 4;
  const int m0 = mt * 64;
  const int np = tid >> 4;        // n-pair 0..15 -> n = 2*np
  const int cg = (tid & 15) * 4;  // column group of 4
  const int mbw = (wave & 3) * 16, nbw = (wave >> 2) * 32;
  float ks = 0.f;
  v8f acc0 = {}, acc1 = {};
  for (int n = 0; n < 4096; n += 32) {
    size_t pr = ((size_t)bh * 4096 + n + 2 * np) * 256 + m0 + cg;
    size_t vr = ((size_t)(b * 4096 + n + 2 * np)) * 1024 + h * 64 + cg;
    v4bf p0 = *(const v4bf*)(PhiK + pr);
    v4bf p1 = *(const v4bf*)(PhiK + pr + 256);
    v4bf v0 = *(const v4bf*)(Vb + vr);
    v4bf v1 = *(const v4bf*)(Vb + vr + 1024);
    if (n + 32 < 4096) {
      __builtin_prefetch(PhiK + pr + 32 * 256, 0, 1);
      __builtin_prefetch(Vb + vr + 32 * 1024, 0, 1);
    }
    __syncthreads();
#pragma unroll
    for (int j = 0; j < 4; ++j) {
      *(unsigned*)&ldsA[(cg + j) * 32 + 2 * np] = pack2(p0[j], p1[j]);
      *(unsigned*)&ldsB[(cg + j) * 32 + 2 * np] = pack2(v0[j], v1[j]);
    }
    __syncthreads();
    if (tid < 64) {
      float s = 0.f;
      for (int nn = 0; nn < 32; ++nn) s += bf16_to_f32(ldsA[tid * 32 + nn]);
      ks += s;
    }
    v16bf a  = frag_a(ldsA, 32, mbw, 0, lane);
    v16bf b0 = frag_b(ldsB, 32, nbw, 0, lane);
    v16bf b1 = frag_b(ldsB, 32, nbw + 16, 0, lane);
    acc0 = wmma_bf16(a, b0, acc0);
    acc1 = wmma_bf16(a, b1, acc1);
  }
  const int col0 = nbw + (lane & 15), col1 = col0 + 16;
  const int row = mbw + ((lane >> 4) ? 8 : 0);
  const size_t kvbase = (size_t)bh * 256 * 64 + (size_t)m0 * 64;
#pragma unroll
  for (int r = 0; r < 8; ++r) {
    KV[kvbase + (size_t)(row + r) * 64 + col0] = acc0[r];
    KV[kvbase + (size_t)(row + r) * 64 + col1] = acc1[r];
  }
  if (tid < 64) Ksum[(size_t)bh * 256 + m0 + tid] = ks;
}

// ---------------------------------------------------------------------------
// out kernel: out[n,d] = (phi_q[n,:] @ kv[:,d]) * z[n],
// z = 1/(phi_q[n,:].k_sum + 1e-6). phi_q tile TDM-loaded; kv^T thread-staged.
// ---------------------------------------------------------------------------
__global__ __launch_bounds__(256) void out_kernel(
    const bf16* __restrict__ PhiQ,  // [BH*N, 256]
    const float* __restrict__ KV,   // [BH, 256, 64]
    const float* __restrict__ Ksum, // [BH, 256]
    bf16* __restrict__ Ot) {        // [B*N, 1024]
  __shared__ bf16 ldsA[32 * 256];   // phi_q tile [n][m]
  __shared__ bf16 ldsKV[64 * 256];  // kv^T [d][m]
  __shared__ float ldsKs[256];
  __shared__ float ldsZ[32];
  const int tid = threadIdx.x, lane = tid & 31, wave = tid >> 5;
  const int bid = blockIdx.x;
  const int nt = bid & 127, h = (bid >> 7) & 15, b = bid >> 11;
  const int bh = b * 16 + h;
  const int n0 = nt * 32;
  if (wave == 0)  // phi_q rows are contiguous (full 256-wide rows)
    tdm_load_tile_2d(lds_off_of(&ldsA[0]),
                     PhiQ + ((size_t)bh * 4096 + n0) * 256,
                     256, 32, 256, 131072, 256);
  for (int i = tid; i < 128 * 64; i += 256) {  // pair-packed kv^T staging
    int mp = i >> 6, d = i & 63;
    float f0 = KV[(size_t)bh * 16384 + (size_t)(2 * mp) * 64 + d];
    float f1 = KV[(size_t)bh * 16384 + (size_t)(2 * mp + 1) * 64 + d];
    *(unsigned*)&ldsKV[d * 256 + 2 * mp] =
        pack2(f32_to_bf16(f0), f32_to_bf16(f1));
  }
  ldsKs[tid] = Ksum[(size_t)bh * 256 + tid];
  if (wave == 0) __builtin_amdgcn_s_wait_tensorcnt(0);
  __syncthreads();
  if (tid < 32) {
    float s = 0.f;
    for (int m = 0; m < 256; ++m)
      s += bf16_to_f32(ldsA[tid * 256 + m]) * ldsKs[m];
    ldsZ[tid] = 1.0f / (s + 1e-6f);
  }
  __syncthreads();
  const int mb = (wave & 1) * 16, nb = (wave >> 1) * 16;
  v8f acc = {};
#pragma unroll
  for (int k0 = 0; k0 < 256; k0 += 32) {
    v16bf a = frag_a(ldsA, 256, mb, k0, lane);
    v16bf bb = frag_b(ldsKV, 256, nb, k0, lane);
    acc = wmma_bf16(a, bb, acc);
  }
  const int col = nb + (lane & 15);
  const int row = mb + ((lane >> 4) ? 8 : 0);
#pragma unroll
  for (int r = 0; r < 8; ++r) {
    float z = ldsZ[row + r];
    size_t o = (size_t)(b * 4096 + n0 + row + r) * 1024 + h * 64;
    Ot[o + col] = f32_to_bf16(acc[r] * z);
  }
}

// ---------------------------------------------------------------------------
// Host orchestration
// ---------------------------------------------------------------------------
extern "C" void kernel_launch(void* const* d_in, const int* in_sizes, int n_in,
                              void* d_out, int out_size, void* d_ws,
                              size_t ws_size, hipStream_t stream) {
  (void)in_sizes; (void)n_in; (void)out_size; (void)ws_size;
  const float* x   = (const float*)d_in[0];
  const float* w_q = (const float*)d_in[1];
  const float* w_k = (const float*)d_in[2];
  const float* w_v = (const float*)d_in[3];
  const float* w_o = (const float*)d_in[4];
  const float* b_o = (const float*)d_in[5];
  const float* om  = (const float*)d_in[6];
  float* out = (float*)d_out;

  char* p = (char*)d_ws;
  auto alloc = [&](size_t bytes) {
    char* r = p;
    p += (bytes + 255) & ~(size_t)255;
    return r;
  };
  const size_t BN = 8192, DM = 1024;
  bf16* xb   = (bf16*)alloc(BN * DM * 2);
  bf16* wqt  = (bf16*)alloc(DM * DM * 2);  // transposed weights [N][K]
  bf16* wkt  = (bf16*)alloc(DM * DM * 2);
  bf16* wvt  = (bf16*)alloc(DM * DM * 2);
  bf16* wot  = (bf16*)alloc(DM * DM * 2);
  bf16* omb  = (bf16*)alloc(256 * 64 * 2);
  bf16* qb   = (bf16*)alloc(BN * DM * 2);
  bf16* kb   = (bf16*)alloc(BN * DM * 2);
  bf16* vb   = (bf16*)alloc(BN * DM * 2);
  bf16* phiq = (bf16*)alloc((size_t)32 * 4096 * 256 * 2);
  bf16* phik = (bf16*)alloc((size_t)32 * 4096 * 256 * 2);
  float* kv   = (float*)alloc((size_t)32 * 256 * 64 * 4);
  float* ksum = (float*)alloc((size_t)32 * 256 * 4);
  bf16* ot   = (bf16*)alloc(BN * DM * 2);

  cast_f32_bf16_kernel<<<dim3((unsigned)((BN * DM + 1023) / 1024)), 256, 0,
                         stream>>>(x, xb, (int)(BN * DM));
  cast_f32_bf16_kernel<<<dim3((256 * 64 + 1023) / 1024), 256, 0, stream>>>(
      om, omb, 256 * 64);
  dim3 gT(32, 32);  // 1024/32 x 1024/32
  transpose_cast_kernel<<<gT, 256, 0, stream>>>(w_q, wqt, 1024, 1024);
  transpose_cast_kernel<<<gT, 256, 0, stream>>>(w_k, wkt, 1024, 1024);
  transpose_cast_kernel<<<gT, 256, 0, stream>>>(w_v, wvt, 1024, 1024);
  transpose_cast_kernel<<<gT, 256, 0, stream>>>(w_o, wot, 1024, 1024);

  dim3 gProj(16, 64);  // N/64=16, M/128=64
  gemm_bf16_kernel<<<gProj, 256, 0, stream>>>(xb, wqt, 8192, 1024, 1024,
                                              nullptr, qb, nullptr);
  gemm_bf16_kernel<<<gProj, 256, 0, stream>>>(xb, wkt, 8192, 1024, 1024,
                                              nullptr, kb, nullptr);
  gemm_bf16_kernel<<<gProj, 256, 0, stream>>>(xb, wvt, 8192, 1024, 1024,
                                              nullptr, vb, nullptr);

  phi_kernel<<<2048, 256, 0, stream>>>(qb, omb, phiq);
  phi_kernel<<<2048, 256, 0, stream>>>(kb, omb, phik);

  kv_kernel<<<128, 256, 0, stream>>>(phik, vb, kv, ksum);

  out_kernel<<<4096, 256, 0, stream>>>(phiq, kv, ksum, ot);

  gemm_bf16_kernel<<<gProj, 256, 0, stream>>>(ot, wot, 8192, 1024, 1024, out,
                                              nullptr, b_o);
}
